// FEM_33251636806316
// MI455X (gfx1250) — matrix operand order repaired and checked by reading, not compiled, not run on hardware
//
#include <hip/hip_runtime.h>
#include <math.h>

// ---------------- CDNA5 WMMA / TDM types ----------------
typedef __attribute__((ext_vector_type(16))) __bf16 v16bf;
typedef __attribute__((ext_vector_type(8)))  __bf16 v8bf;
typedef __attribute__((ext_vector_type(8)))  float  v8f;
typedef __attribute__((ext_vector_type(4)))  unsigned int v4u;
typedef __attribute__((ext_vector_type(8)))  int v8i;
typedef __attribute__((ext_vector_type(4)))  int v4i;

constexpr int TBM = 128;  // tile rows
constexpr int TBN = 64;   // tile cols
constexpr int TBK = 32;   // k-step (bf16 WMMA K)
constexpr int NTHR = 256; // 8 waves

__device__ __forceinline__ unsigned short f2bf(float f) {
  union { float f; unsigned int u; } a; a.f = f;
  return (unsigned short)((a.u + 0x7FFFu + ((a.u >> 16) & 1u)) >> 16);
}
__device__ __forceinline__ void st4bf(unsigned short* dst, float4 v) {
  union { unsigned short u[4]; uint2 q; } t;
  t.u[0] = f2bf(v.x); t.u[1] = f2bf(v.y); t.u[2] = f2bf(v.z); t.u[3] = f2bf(v.w);
  *(uint2*)dst = t.q;   // 8B aligned (k offsets are multiples of 4 halfs)
}

// Tensor Data Mover: load a 2D f32 tile (tile_h rows x tile_w elems, row stride
// in elements) from global into LDS at byte offset lds_off. D# built per CDNA5
// ISA 8.3/8.4 (group0: count=1, lds_addr, global_addr[56:0], type=2;
// group1: data_size=2 (4B), tensor dims, tile dims, dim0 stride; groups2/3 = 0
// for a 2D tensor). Tracked by TENSORcnt. clang-23 6-arg builtin form.
__device__ __forceinline__ void tdm_load_2d_f32(const float* gptr, unsigned lds_off,
                                                unsigned tile_w, unsigned tile_h,
                                                unsigned row_stride) {
  unsigned long long ga = (unsigned long long)(size_t)gptr;
  v4u g0;
  g0[0] = 1u;                                    // count=1 (user descriptor)
  g0[1] = lds_off;                               // lds_addr (bytes)
  g0[2] = (unsigned)(ga & 0xFFFFFFFFu);          // global_addr[31:0]
  g0[3] = (unsigned)((ga >> 32) & 0x01FFFFFFu)   // global_addr[56:32]
        | (2u << 30);                            // type = 2 ("image")
  v8i g1;
  g1[0] = (int)(2u << 16);                       // data_size=4B, no multicast
  g1[1] = (int)((tile_w & 0xFFFFu) << 16);       // tensor_dim0[15:0]  (== tile_w)
  g1[2] = (int)(((tile_w >> 16) & 0xFFFFu) | ((tile_h & 0xFFFFu) << 16)); // dim0 hi | dim1 lo
  g1[3] = (int)(((tile_h >> 16) & 0xFFFFu) | ((tile_w & 0xFFFFu) << 16)); // dim1 hi | tile_dim0
  g1[4] = (int)(tile_h & 0xFFFFu);               // tile_dim1 (tile_dim2 = 0)
  g1[5] = (int)row_stride;                       // tensor_dim0_stride[31:0]
  g1[6] = 0;                                     // stride hi | dim1_stride lo
  g1[7] = 0;
  const v4i gz  = {0, 0, 0, 0};
  const v8i gz8 = {0, 0, 0, 0, 0, 0, 0, 0};
  __builtin_amdgcn_tensor_load_to_lds(g0, g1, gz, gz, gz8, 0);
}

struct GemmP {
  const float* A;  const float* A2; const float* Bm; const float* B2;
  const float* bias;    // [N]
  const float* cscale;  // [M] conv BN scale
  const float* cbias;   // [M] conv BN bias
  const float* res;     // residual (layout of C)
  const float* rowsub;  // [M] per-row subtract (FAVOR+ xd)
  const float* rowdiv;  // per-row divisor (D)
  float* C;
  int M, N, K, lda, ldb, ldc;
  int split, T, KA2;
  float alpha;
  long long bsA, bsB, bsB2, bsC, bsRes, bsRow;
};

// AMODE: 0=row-major [M,K] (TDM fast path), 1=transposed [K,M], 2=token concat
// BMODE: 0=row-major [K,N], 1=transposed [N,K], 2=im2col 3x3 pad1 on 28x28 + concat split
// EPI:   0=none 1=bias 2=bias+gelu(exact) 3=scale*x+bias,relu 4=exp(x-rowsub)*alpha
//        5=bias+residual 6=x/(rowdiv+1e-8)
template<int AMODE, int BMODE, int EPI>
__global__ __launch_bounds__(NTHR)
void gemm_wmma(GemmP p) {
  __shared__ __align__(32) unsigned short As[TBM * TBK];  // bf16 [m][k]
  __shared__ __align__(32) unsigned short Bs[TBN * TBK];  // bf16 [n][k]
  __shared__ __align__(16) float Astage[TBM * TBK];       // TDM f32 staging
  const int tid  = threadIdx.x;
  const int wv   = tid >> 5;
  const int lane = tid & 31;
  const int g    = lane >> 4;
  const int ml   = lane & 15;
  const int bz   = blockIdx.z;
  const int m0   = blockIdx.y * TBM;
  const int n0   = blockIdx.x * TBN;

  const float* Ap  = p.A  + (long long)bz * p.bsA;
  const float* Bp  = p.Bm + (long long)bz * p.bsB;
  const float* B2p = p.B2 ? (p.B2 + (long long)bz * p.bsB2) : Bp;
  float*       Cp  = p.C  + (long long)bz * p.bsC;
  const unsigned astage_off = (unsigned)(size_t)(void*)&Astage[0];

  const v8f vzero = {0.f,0.f,0.f,0.f,0.f,0.f,0.f,0.f};
  v8f acc[4]; acc[0]=vzero; acc[1]=vzero; acc[2]=vzero; acc[3]=vzero;

  for (int k0 = 0; k0 < p.K; k0 += TBK) {
    const bool fullK = (k0 + TBK <= p.K);
    const bool tdmA  = (AMODE == 0) && fullK && (m0 + TBM <= p.M);
    if (BMODE == 0) {  // gfx1250 global_prefetch of next weight tile
      int nk = k0 + TBK;
      if (nk < p.K) __builtin_prefetch(&Bp[(long long)nk * p.ldb + n0], 0, 1);
    }

    // ============ phase A: stage A tile (TDM) / load A+B tiles ============
    if (tdmA) {
      if (wv == 0) {  // one wave drives the Tensor Data Mover
        tdm_load_2d_f32(Ap + (long long)m0 * p.lda + k0, astage_off,
                        (unsigned)TBK, (unsigned)TBM, (unsigned)p.lda);
        __builtin_amdgcn_s_wait_tensorcnt(0);
      }
    } else if (AMODE == 0 || AMODE == 2) {
      const bool vecA = (AMODE == 2) && fullK && (m0 + TBM <= p.M) &&
                        ((k0 + TBK) <= p.split);
      if (vecA) {
        #pragma unroll
        for (int u = tid; u < (TBM*TBK)/4; u += NTHR) {
          int mr = u >> 3;            // 8 float4 per 32-wide row
          int kk = (u & 7) * 4;
          const float4 v4 = *(const float4*)(Ap + (long long)(m0+mr)*p.lda + (k0+kk));
          st4bf(&As[mr*TBK + kk], v4);
        }
      } else if (AMODE == 2 && fullK && k0 >= p.split) {
        // token-concat tail: x3 stored [B, KA2, T] (channel-major gather)
        for (int i = tid; i < TBM*TBK; i += NTHR) {
          int mr = i >> 5, kk = i & (TBK-1);
          int m = m0 + mr, k = k0 + kk;
          int bb = m / p.T, tt = m - bb * p.T;
          As[i] = f2bf(p.A2[((long long)bb * p.KA2 + (k - p.split)) * p.T + tt]);
        }
      } else {
        for (int i = tid; i < TBM*TBK; i += NTHR) {
          int mr = i >> 5, kk = i & (TBK-1);
          int m = m0 + mr, k = k0 + kk;
          float v = 0.f;
          if (m < p.M && k < p.K) {
            if (AMODE == 0 || k < p.split) v = Ap[(long long)m * p.lda + k];
            else {
              int bb = m / p.T, tt = m - bb * p.T;
              v = p.A2[((long long)bb * p.KA2 + (k - p.split)) * p.T + tt];
            }
          }
          As[i] = f2bf(v);
        }
      }
    } else { // AMODE == 1 : A[m,k] = Ap[k*lda + m]
      if (fullK && (m0 + TBM <= p.M)) {
        #pragma unroll
        for (int u = tid; u < (TBM*TBK)/4; u += NTHR) {
          int kk = u & (TBK-1);
          int mr = (u >> 5) * 4;
          const float4 v4 = *(const float4*)(Ap + (long long)(k0+kk)*p.lda + (m0+mr));
          As[(mr+0)*TBK + kk] = f2bf(v4.x);
          As[(mr+1)*TBK + kk] = f2bf(v4.y);
          As[(mr+2)*TBK + kk] = f2bf(v4.z);
          As[(mr+3)*TBK + kk] = f2bf(v4.w);
        }
      } else {
        for (int i = tid; i < TBM*TBK; i += NTHR) {
          int mr = i >> 5, kk = i & (TBK-1);
          int m = m0 + mr, k = k0 + kk;
          float v = 0.f;
          if (m < p.M && k < p.K) v = Ap[(long long)k * p.lda + m];
          As[i] = f2bf(v);
        }
      }
    }

    // ---------------- B tile (32x64) -> bf16 LDS [n][k] ----------------
    if (BMODE == 0) {
      if (fullK && (n0 + TBN <= p.N)) {
        #pragma unroll
        for (int u = tid; u < (TBK*TBN)/4; u += NTHR) {
          int kr = u >> 4;            // 16 float4 per 64-wide row
          int nn = (u & 15) * 4;
          const float4 v4 = *(const float4*)(Bp + (long long)(k0+kr)*p.ldb + (n0+nn));
          Bs[(nn+0)*TBK + kr] = f2bf(v4.x);
          Bs[(nn+1)*TBK + kr] = f2bf(v4.y);
          Bs[(nn+2)*TBK + kr] = f2bf(v4.z);
          Bs[(nn+3)*TBK + kr] = f2bf(v4.w);
        }
      } else {
        for (int i = tid; i < TBK*TBN; i += NTHR) {
          int kr = i >> 6, nn = i & (TBN-1);
          int k = k0 + kr, n = n0 + nn;
          float v = 0.f;
          if (n < p.N && k < p.K) v = Bp[(long long)k * p.ldb + n];
          Bs[nn * TBK + kr] = f2bf(v);
        }
      }
    } else if (BMODE == 1) { // B[k,n] = Bp[n*ldb + k]
      if (fullK && (n0 + TBN <= p.N)) {
        #pragma unroll
        for (int u = tid; u < (TBK*TBN)/4; u += NTHR) {
          int nn = u >> 3;            // 8 float4 per 32-wide k row
          int kk = (u & 7) * 4;
          const float4 v4 = *(const float4*)(Bp + (long long)(n0+nn)*p.ldb + (k0+kk));
          st4bf(&Bs[nn*TBK + kk], v4);
        }
      } else {
        for (int i = tid; i < TBK*TBN; i += NTHR) {
          int kr = i >> 6, nn = i & (TBN-1);
          int k = k0 + kr, n = n0 + nn;
          float v = 0.f;
          if (n < p.N && k < p.K) v = Bp[(long long)n * p.ldb + k];
          Bs[nn * TBK + kr] = f2bf(v);
        }
      }
    } else { // BMODE == 2 : implicit im2col, one k-row per thread, 8 positions each
      int kr = tid >> 3;
      int k  = k0 + kr;
      int nb = (tid & 7) * 8;
      if (k < p.K) {
        int ci = k / 9, rs = k - ci * 9;
        int rr = rs / 3, ss = rs - rr * 3;
        const float* src = (ci < p.split) ? (Bp  + (long long)ci * 784)
                                          : (B2p + (long long)(ci - p.split) * 784);
        #pragma unroll
        for (int j = 0; j < 8; ++j) {
          int nn = nb + j, n = n0 + nn;
          float v = 0.f;
          if (n < p.N) {
            int hh = n / 28, ww = n - hh * 28;
            int ih = hh + rr - 1, iw = ww + ss - 1;
            if (ih >= 0 && ih < 28 && iw >= 0 && iw < 28) v = src[ih * 28 + iw];
          }
          Bs[nn * TBK + kr] = f2bf(v);
        }
      } else {
        #pragma unroll
        for (int j = 0; j < 8; ++j) Bs[(nb + j) * TBK + kr] = 0;
      }
    }
    __syncthreads();

    // ============ phase B: all waves pack TDM-staged f32 -> bf16 ============
    if (tdmA) {
      #pragma unroll
      for (int u = tid; u < (TBM*TBK)/4; u += NTHR) {
        const float4 v4 = *(const float4*)&Astage[u * 4];
        st4bf(&As[u * 4], v4);
      }
    }
    __syncthreads();

    // ============ phase C: fragments + WMMA ============
    // A fragment: lane-group g holds K range [16g,16g+16)
    union { v16bf v; struct { v8bf lo, hi; } h; } afrag;
    const unsigned short* arow = &As[(wv * 16 + ml) * TBK];
    afrag.h.lo = *(const v8bf*)(arow + 16 * g);
    afrag.h.hi = *(const v8bf*)(arow + 16 * g + 8);

    #pragma unroll
    for (int sn = 0; sn < 4; ++sn) {
      // B fragment: lane holds column n=ml of subtile sn, K range [16g,16g+16)
      const v16bf bfrag = *(const v16bf*)&Bs[(sn * 16 + ml) * TBK + 16 * g];
      acc[sn] = __builtin_amdgcn_wmma_f32_16x16x32_bf16(
          false, afrag.v, false, bfrag, (short)0, acc[sn], false, false);
    }
    __syncthreads();
  }

  const float* resp = p.res    ? (p.res    + (long long)bz * p.bsRes) : nullptr;
  const float* rdp  = p.rowdiv ? (p.rowdiv + (long long)bz * p.bsRow) : nullptr;
  #pragma unroll
  for (int sn = 0; sn < 4; ++sn) {
    int n = n0 + sn * 16 + ml;
    if (n >= p.N) continue;
    #pragma unroll
    for (int e = 0; e < 8; ++e) {
      int m = m0 + wv * 16 + e + 8 * g;   // D layout: VGPR e, lane-group g
      if (m >= p.M) continue;
      float v = acc[sn][e];
      if (EPI == 1)      { v += p.bias[n]; }
      else if (EPI == 2) { v += p.bias[n];
                           v = 0.5f * v * (1.0f + erff(v * 0.70710678118654752f)); }
      else if (EPI == 3) { v = v * p.cscale[m] + p.cbias[m]; v = fmaxf(v, 0.f); }
      else if (EPI == 4) { v = expf(v - p.rowsub[m]) * p.alpha; }
      else if (EPI == 5) { v = v + p.bias[n] + resp[(long long)m * p.ldc + n]; }
      else if (EPI == 6) { v = v / (rdp[m] + 1e-8f); }
      Cp[(long long)m * p.ldc + n] = v;
    }
  }
}

// ---------------- small kernels ----------------
__global__ __launch_bounds__(256) void k_pixshuf(const float* bin, float* t) {
  long long idx = (long long)blockIdx.x * 256 + threadIdx.x;
  if (idx >= (long long)64*256*784) return;
  int w2 = idx % 28; long long r = idx / 28;
  int h2 = r % 28; r /= 28;
  int ch = r % 256; int b = (int)(r / 256);
  int hh = h2 >> 1, ii = h2 & 1, ww = w2 >> 1, jj = w2 & 1;
  t[idx] = bin[((long long)b*196 + (hh*14 + ww))*1024 + ch*4 + ii*2 + jj];
}

__global__ __launch_bounds__(256) void k_samm(const float* t, float* sain) {
  int idx = blockIdx.x * 256 + threadIdx.x;
  if (idx >= 64*784) return;
  int b = idx / 784, pq = idx - b*784;
  const float* base = t + (long long)b*256*784 + pq;
  float s = 0.f, mx = -3.4e38f;
  for (int ch = 0; ch < 256; ++ch) { float v = base[(long long)ch*784]; s += v; mx = fmaxf(mx, v); }
  sain[(long long)b*1568 + pq]       = s * (1.f/256.f);
  sain[(long long)b*1568 + 784 + pq] = mx;
}

__global__ __launch_bounds__(256) void k_saconv(const float* sain, const float* saw, float* ts) {
  int idx = blockIdx.x * 256 + threadIdx.x;
  if (idx >= 64*784) return;
  int b = idx / 784, pq = idx - b*784;
  int hh = pq / 28, ww = pq - hh*28;
  float acc = 0.f;
  for (int ci = 0; ci < 2; ++ci)
    for (int r = 0; r < 7; ++r) {
      int ih = hh + r - 3; if (ih < 0 || ih >= 28) continue;
      for (int s = 0; s < 7; ++s) {
        int iw = ww + s - 3; if (iw < 0 || iw >= 28) continue;
        acc += sain[(long long)b*1568 + ci*784 + ih*28 + iw] * saw[ci*49 + r*7 + s];
      }
    }
  ts[idx] = 1.f / (1.f + expf(-acc));
}

__global__ __launch_bounds__(256) void k_ca(const float* c, const float* w1, const float* w2,
                                            float* cc) {
  __shared__ float smean[256], smax[256], hm[16], hx[16];
  int b = blockIdx.x, ch = threadIdx.x;
  const float* base = c + ((long long)b*256 + ch)*784;
  float s = 0.f, mx = -3.4e38f;
  for (int pq = 0; pq < 784; ++pq) { float v = base[pq]; s += v; mx = fmaxf(mx, v); }
  smean[ch] = s * (1.f/784.f); smax[ch] = mx;
  __syncthreads();
  if (ch < 16) {
    float am = 0.f, ax = 0.f;
    for (int i = 0; i < 256; ++i) { float w = w1[i*16 + ch]; am += smean[i]*w; ax += smax[i]*w; }
    hm[ch] = fmaxf(am, 0.f); hx[ch] = fmaxf(ax, 0.f);
  }
  __syncthreads();
  float o = 0.f;
  for (int j = 0; j < 16; ++j) o += (hm[j] + hx[j]) * w2[j*256 + ch];
  cc[b*256 + ch] = 1.f / (1.f + expf(-o));
}

__global__ __launch_bounds__(256) void k_txcx(const float* t, const float* c, const float* cc,
                                              const float* ts, float* ob) {
  long long idx = (long long)blockIdx.x * 256 + threadIdx.x;
  if (idx >= (long long)64*512*784) return;
  int pq = idx % 784; long long r = idx / 784;
  int ch = r % 512; int b = (int)(r / 512);
  float v;
  if (ch < 256) v = t[((long long)b*256 + ch)*784 + pq] * cc[b*256 + ch];
  else          v = c[((long long)b*256 + (ch-256))*784 + pq] * ts[b*784 + pq];
  ob[idx] = v;
}

__global__ __launch_bounds__(256) void k_ln(const float* x, const float* g, const float* bt,
                                            float* yo, int R) {
  int wv = threadIdx.x >> 5, lane = threadIdx.x & 31;
  int r = blockIdx.x * 8 + wv;
  if (r >= R) return;
  const float* xr = x + (long long)r * 512;
  float v[16]; float s = 0.f;
  #pragma unroll
  for (int e = 0; e < 16; ++e) { v[e] = xr[lane + 32*e]; s += v[e]; }
  #pragma unroll
  for (int o = 16; o > 0; o >>= 1) s += __shfl_xor(s, o);
  float mu = s * (1.f/512.f);
  float q = 0.f;
  #pragma unroll
  for (int e = 0; e < 16; ++e) { float d = v[e]-mu; q += d*d; }
  #pragma unroll
  for (int o = 16; o > 0; o >>= 1) q += __shfl_xor(q, o);
  float rstd = rsqrtf(q * (1.f/512.f) + 1e-5f);
  float* yr = yo + (long long)r * 512;
  #pragma unroll
  for (int e = 0; e < 16; ++e) { int ci = lane + 32*e; yr[ci] = (v[e]-mu)*rstd*g[ci] + bt[ci]; }
}

__global__ __launch_bounds__(256) void k_xd(const float* kqv, float* xdk, float* xdq, int R) {
  int wv = threadIdx.x >> 5, lane = threadIdx.x & 31;
  int r = blockIdx.x * 8 + wv;
  if (r >= R) return;
  const float* kr = kqv + (long long)r * 1536;
  float sk = 0.f, sq = 0.f;
  #pragma unroll
  for (int e = 0; e < 16; ++e) {
    float a = kr[lane + 32*e];        sk += a*a;
    float b = kr[512 + lane + 32*e];  sq += b*b;
  }
  #pragma unroll
  for (int o = 16; o > 0; o >>= 1) { sk += __shfl_xor(sk, o); sq += __shfl_xor(sq, o); }
  if (lane == 0) { xdk[r] = 0.5f*sk; xdq[r] = 0.5f*sq; }
}

__global__ __launch_bounds__(256) void k_kpsum(const float* kp, float* kps) {
  int b = blockIdx.x, m = threadIdx.x;
  const float* base = kp + (long long)b*784*256 + m;
  float s = 0.f;
  for (int t = 0; t < 784; ++t) s += base[(long long)t*256];
  kps[b*256 + m] = s;
}

__global__ __launch_bounds__(256) void k_Dvec(const float* qp, const float* kps, float* Dv, int R) {
  int wv = threadIdx.x >> 5, lane = threadIdx.x & 31;
  int r = blockIdx.x * 8 + wv;
  if (r >= R) return;
  int b = r / 784;
  float s = 0.f;
  #pragma unroll
  for (int e = 0; e < 8; ++e) { int m = lane + 32*e; s += qp[(long long)r*256 + m] * kps[b*256 + m]; }
  #pragma unroll
  for (int o = 16; o > 0; o >>= 1) s += __shfl_xor(s, o);
  if (lane == 0) Dv[r] = s;
}

// ---------------- host orchestration ----------------
extern "C" void kernel_launch(void* const* d_in, const int* in_sizes, int n_in,
                              void* d_out, int out_size, void* d_ws, size_t ws_size,
                              hipStream_t stream) {
  (void)in_sizes; (void)n_in; (void)out_size; (void)ws_size;
  const float* a      = (const float*)d_in[0];
  const float* bin    = (const float*)d_in[1];
  const float* cin    = (const float*)d_in[2];
  const float* cf_w1  = (const float*)d_in[3];
  const float* cf_b1  = (const float*)d_in[4];
  const float* cf_w2  = (const float*)d_in[5];
  const float* cf_b2  = (const float*)d_in[6];
  const float* ln1_g  = (const float*)d_in[7];
  const float* ln1_b  = (const float*)d_in[8];
  const float* kqv_w  = (const float*)d_in[9];
  const float* kqv_b  = (const float*)d_in[10];
  const float* proj_w = (const float*)d_in[11];
  const float* proj_b = (const float*)d_in[12];
  const float* ln2_g  = (const float*)d_in[13];
  const float* ln2_b  = (const float*)d_in[14];
  const float* mlp_w1 = (const float*)d_in[15];
  const float* mlp_b1 = (const float*)d_in[16];
  const float* mlp_w2 = (const float*)d_in[17];
  const float* mlp_b2 = (const float*)d_in[18];
  const float* ca_w1  = (const float*)d_in[19];
  const float* ca_w2  = (const float*)d_in[20];
  const float* sa_w   = (const float*)d_in[21];
  const float* cw[5]; const float* cs[5]; const float* cb[5];
  for (int i = 0; i < 5; ++i) {
    cw[i] = (const float*)d_in[22 + 3*i];
    cs[i] = (const float*)d_in[23 + 3*i];
    cb[i] = (const float*)d_in[24 + 3*i];
  }
  const float* wrf = (const float*)d_in[37];
  float* out = (float*)d_out;

  const int R = 64 * 784;                 // 50176 tokens
  const size_t SP  = (size_t)64*256*784;  // [B,256,28,28]
  const size_t TOK = (size_t)64*784*512;  // [B,T,512]
  float* W = (float*)d_ws; size_t off = 0;
  auto alloc = [&](size_t n) { float* p = W + off; off += n; return p; };
  float* t    = alloc(SP);
  float* sain = alloc((size_t)64*2*784);
  float* ts   = alloc((size_t)64*784);
  float* cc   = alloc((size_t)64*256);
  float* txcx = alloc(2*SP);
  float* x1   = alloc(SP);
  float* tx   = alloc(SP);
  float* cx   = alloc(SP);
  float* x2   = alloc(SP);
  float* x3   = alloc(SP);
  float* zmid = alloc(TOK);
  float* z    = alloc(TOK);
  float* h    = alloc(TOK);
  float* kqv  = alloc((size_t)64*784*1536);
  float* xdk  = alloc((size_t)64*784);
  float* xdq  = alloc((size_t)64*784);
  float* kp   = alloc((size_t)64*784*256);
  float* qp   = alloc((size_t)64*784*256);
  float* kps  = alloc((size_t)64*256);
  float* Dv   = alloc((size_t)64*784);
  float* kptv = alloc((size_t)64*512*256);
  float* y    = alloc(TOK);

  // ---- FEM front end ----
  k_pixshuf<<<dim3((unsigned)(SP/256)), 256, 0, stream>>>(bin, t);
  k_samm  <<<196, 256, 0, stream>>>(t, sain);
  k_saconv<<<196, 256, 0, stream>>>(sain, sa_w, ts);
  k_ca    <<<64, 256, 0, stream>>>(cin, ca_w1, ca_w2, cc);
  k_txcx  <<<dim3((unsigned)(2*SP/256)), 256, 0, stream>>>(t, cin, cc, ts, txcx);

  // ---- conv stack: implicit-GEMM WMMA, fused BN+ReLU ----
  auto conv = [&](const float* wgt, const float* sc, const float* bi,
                  const float* s0, long long bs0, const float* s1, int split, int K, float* dst) {
    GemmP p{};
    p.A = wgt; p.Bm = s0; p.B2 = s1;
    p.cscale = sc; p.cbias = bi; p.C = dst;
    p.M = 256; p.N = 784; p.K = K;
    p.lda = K; p.ldc = 784; p.split = split;
    p.bsB = bs0; p.bsB2 = (long long)256*784; p.bsC = (long long)256*784;
    gemm_wmma<0,2,3><<<dim3(13, 2, 64), NTHR, 0, stream>>>(p);
  };
  conv(cw[0], cs[0], cb[0], txcx, (long long)512*784, nullptr, 512, 4608, x1);
  conv(cw[1], cs[1], cb[1], t,    (long long)256*784, x1,      256, 4608, tx);
  conv(cw[2], cs[2], cb[2], cin,  (long long)256*784, x1,      256, 4608, cx);
  conv(cw[3], cs[3], cb[3], tx,   (long long)256*784, cx,      256, 4608, x2);
  conv(cw[4], cs[4], cb[4], x2,   (long long)256*784, nullptr, 256, 2304, x3);

  // ---- conv_fuse: gelu(concat(a, tokens(x3)) @ cf_w1 + b1) @ cf_w2 + b2 ----
  { GemmP p{};
    p.A = a; p.A2 = x3; p.Bm = cf_w1; p.bias = cf_b1; p.C = zmid;
    p.M = R; p.N = 512; p.K = 768; p.lda = 512; p.ldb = 512; p.ldc = 512;
    p.split = 512; p.T = 784; p.KA2 = 256;
    gemm_wmma<2,0,2><<<dim3(8, 392, 1), NTHR, 0, stream>>>(p); }
  { GemmP p{};
    p.A = zmid; p.Bm = cf_w2; p.bias = cf_b2; p.C = z;
    p.M = R; p.N = 512; p.K = 512; p.lda = 512; p.ldb = 512; p.ldc = 512;
    gemm_wmma<0,0,1><<<dim3(8, 392, 1), NTHR, 0, stream>>>(p); }

  // ---- Token performer ----
  k_ln<<<6272, 256, 0, stream>>>(z, ln1_g, ln1_b, h, R);
  { GemmP p{};                                   // kqv = h @ kqv_w + b
    p.A = h; p.Bm = kqv_w; p.bias = kqv_b; p.C = kqv;
    p.M = R; p.N = 1536; p.K = 512; p.lda = 512; p.ldb = 1536; p.ldc = 1536;
    gemm_wmma<0,0,1><<<dim3(24, 392, 1), NTHR, 0, stream>>>(p); }
  k_xd<<<6272, 256, 0, stream>>>(kqv, xdk, xdq, R);
  { GemmP p{};                                   // kp = exp(k@w^T - xdk)/sqrt(M)
    p.A = kqv; p.Bm = wrf; p.rowsub = xdk; p.C = kp;
    p.M = R; p.N = 256; p.K = 512; p.lda = 1536; p.ldb = 512; p.ldc = 256;
    p.alpha = 0.0625f;
    gemm_wmma<0,1,4><<<dim3(4, 392, 1), NTHR, 0, stream>>>(p); }
  { GemmP p{};                                   // qp = exp(q@w^T - xdq)/sqrt(M)
    p.A = kqv + 512; p.Bm = wrf; p.rowsub = xdq; p.C = qp;
    p.M = R; p.N = 256; p.K = 512; p.lda = 1536; p.ldb = 512; p.ldc = 256;
    p.alpha = 0.0625f;
    gemm_wmma<0,1,4><<<dim3(4, 392, 1), NTHR, 0, stream>>>(p); }
  k_kpsum<<<64, 256, 0, stream>>>(kp, kps);
  k_Dvec <<<6272, 256, 0, stream>>>(qp, kps, Dv, R);
  { GemmP p{};                                   // kptv[b] = v^T @ kp  (512x256, K=784)
    p.A = kqv + 1024; p.Bm = kp; p.C = kptv;
    p.M = 512; p.N = 256; p.K = 784; p.lda = 1536; p.ldb = 256; p.ldc = 256;
    p.bsA = (long long)784*1536; p.bsB = (long long)784*256; p.bsC = (long long)512*256;
    gemm_wmma<1,0,0><<<dim3(4, 4, 64), NTHR, 0, stream>>>(p); }
  { GemmP p{};                                   // y[b] = (qp @ kptv^T) / (D+1e-8)
    p.A = qp; p.Bm = kptv; p.rowdiv = Dv; p.C = y;
    p.M = 784; p.N = 512; p.K = 256; p.lda = 256; p.ldb = 256; p.ldc = 512;
    p.bsA = (long long)784*256; p.bsB = (long long)512*256; p.bsC = (long long)784*512;
    p.bsRow = 784;
    gemm_wmma<0,1,6><<<dim3(8, 7, 64), NTHR, 0, stream>>>(p); }
  { GemmP p{};                                   // z += y @ proj_w + proj_b  (in-place)
    p.A = y; p.Bm = proj_w; p.bias = proj_b; p.res = z; p.C = z;
    p.M = R; p.N = 512; p.K = 512; p.lda = 512; p.ldb = 512; p.ldc = 512;
    gemm_wmma<0,0,5><<<dim3(8, 392, 1), NTHR, 0, stream>>>(p); }

  // ---- MLP tail ----
  k_ln<<<6272, 256, 0, stream>>>(z, ln2_g, ln2_b, h, R);
  { GemmP p{};                                   // zmid = gelu(h @ mlp_w1 + b1)
    p.A = h; p.Bm = mlp_w1; p.bias = mlp_b1; p.C = zmid;
    p.M = R; p.N = 512; p.K = 512; p.lda = 512; p.ldb = 512; p.ldc = 512;
    gemm_wmma<0,0,2><<<dim3(8, 392, 1), NTHR, 0, stream>>>(p); }
  { GemmP p{};                                   // out = z + zmid @ mlp_w2 + b2
    p.A = zmid; p.Bm = mlp_w2; p.bias = mlp_b2; p.res = z; p.C = out;
    p.M = R; p.N = 512; p.K = 512; p.lda = 512; p.ldb = 512; p.ldc = 512;
    gemm_wmma<0,0,5><<<dim3(8, 392, 1), NTHR, 0, stream>>>(p); }
}
